// MixtureOfExperts_64183991271494
// MI455X (gfx1250) — matrix operand order repaired and checked by reading, not compiled, not run on hardware
//
#include <hip/hip_runtime.h>
#include <hip/hip_bf16.h>

// MoE forward for MI455X (gfx1250, wave32).
// Strategy: top-2 routing (8x FLOP cut vs dense reference), bf16 WMMA with
// f32 accumulation, weights/activations converted once into L2-resident
// bf16 scratch, fused FFN (x@W1 -> relu -> @W2) per expert tile.
// Round 3 fix: epilogue mt-loop must be FULLY unrolled -- with "#pragma
// unroll 1" the register arrays hacc[mt][t] get dynamically indexed and the
// compiler lowers them to scratch (the 16x scratch_store_b128 seen in round
// 2). Keep the k0 loop rolled (that fixed the round-1 overspill).

#define B_TOK   16384
#define D_IN    512
#define H_DIM   1024
#define N_EXP   16
#define O_DIM   10
#define TPB_TOK 64   // tokens per block in the expert kernel (4 M-tiles of 16)

typedef unsigned short u16;
typedef __attribute__((ext_vector_type(8)))  __bf16 v8bf;
typedef __attribute__((ext_vector_type(16))) __bf16 v16bf;
typedef __attribute__((ext_vector_type(8)))  float  v8f;

__device__ __forceinline__ u16 f2bf(float f) {
  union { float f; unsigned u; } cv; cv.f = f;
  unsigned u = cv.u;
  return (u16)((u + 0x7FFFu + ((u >> 16) & 1u)) >> 16);  // round-nearest-even
}

// Load 16 bf16 elements as two 16-byte chunks: p[0..7] and p[gap..gap+7].
// gap=16 -> K interleave 0-7 / 16-23 per half-wave (AMD table order; unused here)
// gap=8  -> 16 contiguous K per lane (k = 16*half + s), used here for both A and B
__device__ __forceinline__ v16bf ld16(const u16* p, int gap) {
  v8bf lo = *(const v8bf*)(p);
  v8bf hi = *(const v8bf*)(p + gap);
  return __builtin_shufflevector(lo, hi, 0,1,2,3,4,5,6,7,8,9,10,11,12,13,14,15);
}

__device__ __forceinline__ v8f zero8() {
  v8f z;
#pragma unroll
  for (int i = 0; i < 8; ++i) z[i] = 0.f;
  return z;
}

// ---------------------------------------------------------------- init
__global__ void init_kernel(float* outW, int* cnt) {
  int i = blockIdx.x * 256 + threadIdx.x;
  if (i < B_TOK * O_DIM) outW[i] = 0.f;
  if (i < N_EXP) cnt[i] = 0;
}

// ---------------------------------------------------------------- converts
__global__ void cvt_x(const float* __restrict__ x, u16* __restrict__ xbf) {
  int i = blockIdx.x * 256 + threadIdx.x;
  if (i < B_TOK * D_IN) xbf[i] = f2bf(x[i]);
}

// W1 [e][k][h] f32  ->  w1t [e][h][k] bf16   (B operand wants K contiguous)
__global__ void cvt_w1(const float* __restrict__ W1, u16* __restrict__ w1t) {
  int i = blockIdx.x * 256 + threadIdx.x;
  if (i >= N_EXP * H_DIM * D_IN) return;
  int e = i >> 19;                 // H_DIM*D_IN = 2^19
  int r = i & ((1 << 19) - 1);
  int h = r >> 9;                  // D_IN = 2^9
  int k = r & (D_IN - 1);
  w1t[i] = f2bf(W1[(e * D_IN + k) * H_DIM + h]);
}

// W2 [e][h][o] f32  ->  w2t [e][n16][h] bf16, padded n=10..15 with zeros
__global__ void cvt_w2(const float* __restrict__ W2, u16* __restrict__ w2t) {
  int i = blockIdx.x * 256 + threadIdx.x;
  if (i >= N_EXP * 16 * H_DIM) return;
  int e = i >> 14;                 // 16*H_DIM = 2^14
  int r = i & ((1 << 14) - 1);
  int n = r >> 10;                 // H_DIM = 2^10
  int h = r & (H_DIM - 1);
  w2t[i] = (n < O_DIM) ? f2bf(W2[(e * H_DIM + h) * O_DIM + n]) : (u16)0;
}

// ---------------------------------------------------------------- gating
// One wave32 per token: logits, softmax, top-2 (jax tie-break: lowest index),
// renormalized gates to d_out, and atomic scatter into per-expert lists.
__global__ void gate_kernel(const float* __restrict__ x, const float* __restrict__ Wg,
                            const float* __restrict__ bg, float* __restrict__ gatesOut,
                            int* __restrict__ cnt, int* __restrict__ lists,
                            float* __restrict__ gwbuf) {
  const int lane = threadIdx.x & 31;
  const int wv   = threadIdx.x >> 5;
  const int b    = blockIdx.x * 8 + wv;

  float xv[16];
#pragma unroll
  for (int i = 0; i < 16; ++i) xv[i] = x[b * D_IN + lane + 32 * i];

  float logit = -__builtin_inff();
  for (int e = 0; e < N_EXP; ++e) {
    float s = 0.f;
#pragma unroll
    for (int i = 0; i < 16; ++i) s += xv[i] * Wg[(lane + 32 * i) * N_EXP + e];
#pragma unroll
    for (int off = 16; off > 0; off >>= 1) s += __shfl_xor(s, off, 32);
    if (lane == e) logit = s + bg[e];
  }

  float mx = logit;
#pragma unroll
  for (int off = 16; off > 0; off >>= 1) mx = fmaxf(mx, __shfl_xor(mx, off, 32));
  float p = (lane < N_EXP) ? __expf(logit - mx) : 0.f;
  float ssum = p;
#pragma unroll
  for (int off = 16; off > 0; off >>= 1) ssum += __shfl_xor(ssum, off, 32);

  // top-1
  float v1 = logit; int i1 = lane;
#pragma unroll
  for (int off = 16; off > 0; off >>= 1) {
    float ov = __shfl_xor(v1, off, 32);
    int   oi = __shfl_xor(i1, off, 32);
    if (ov > v1 || (ov == v1 && oi < i1)) { v1 = ov; i1 = oi; }
  }
  // top-2
  float v2 = (lane == i1) ? -__builtin_inff() : logit; int i2 = lane;
#pragma unroll
  for (int off = 16; off > 0; off >>= 1) {
    float ov = __shfl_xor(v2, off, 32);
    int   oi = __shfl_xor(i2, off, 32);
    if (ov > v2 || (ov == v2 && oi < i2)) { v2 = ov; i2 = oi; }
  }

  float p1 = __shfl(p, i1, 32);
  float p2 = __shfl(p, i2, 32);
  // (p/ssum) / ((p1+p2)/ssum + 1e-12) == p / (p1 + p2 + 1e-12*ssum)
  float denom = p1 + p2 + 1e-12f * ssum;

  if (lane < N_EXP) {
    float g = (lane == i1 || lane == i2) ? (p / denom) : 0.f;
    gatesOut[b * N_EXP + lane] = g;
  }
  if (lane == 0) {
    float g1 = p1 / denom, g2 = p2 / denom;
    int q1 = atomicAdd(cnt + i1, 1);
    lists[i1 * B_TOK + q1] = b;  gwbuf[i1 * B_TOK + q1] = g1;
    int q2 = atomicAdd(cnt + i2, 1);
    lists[i2 * B_TOK + q2] = b;  gwbuf[i2 * B_TOK + q2] = g2;
  }
}

// ---------------------------------------------------------------- expert FFN
// Block = (expert, tile of 64 tokens). 8 waves; wave w owns H columns
// [w*128, w*128+128) in 4 chunks of 32. Per chunk: 16x WMMA K-loop for x@W1,
// relu+b1, LDS bounce to A-layout, 1 WMMA vs W2 (K=32,N=16) into out tile.
__global__ __launch_bounds__(256) void moe_kernel(
    const u16* __restrict__ xbf, const u16* __restrict__ w1t, const u16* __restrict__ w2t,
    const float* __restrict__ b1, const float* __restrict__ b2,
    const int* __restrict__ cnt, const int* __restrict__ lists,
    const float* __restrict__ gwbuf, float* __restrict__ outW) {
  const int e    = blockIdx.y;
  const int tile = blockIdx.x;
  const int count = cnt[e];
  if (tile * TPB_TOK >= count) return;   // uniform early exit

  __shared__ int   toks[TPB_TOK];
  __shared__ float gws[TPB_TOK];
  __shared__ u16   hs[8][16 * 32];       // per-wave relayout buffer (1 KB each)
  __shared__ float red[8 * 4 * 8 * 32];  // cross-wave reduction (32 KB)

  if (threadIdx.x < TPB_TOK) {
    int idx = tile * TPB_TOK + threadIdx.x;
    if (idx < count) {
      toks[threadIdx.x] = lists[e * B_TOK + idx];
      gws[threadIdx.x]  = gwbuf[e * B_TOK + idx];
    } else {
      toks[threadIdx.x] = 0;   // valid row, contribution killed by weight 0
      gws[threadIdx.x]  = 0.f;
    }
  }
  __syncthreads();

  const int lane = threadIdx.x & 31;
  const int wv   = threadIdx.x >> 5;
  const int m    = lane & 15;   // row (A) or column (B/C)
  const int kg   = lane >> 4;   // half-wave K group

  const u16* w1e = w1t + e * (H_DIM * D_IN);
  const u16* w2e = w2t + e * (16 * H_DIM);

  const u16* xrow[4];
#pragma unroll
  for (int mt = 0; mt < 4; ++mt) xrow[mt] = xbf + toks[mt * 16 + m] * D_IN;

  v8f outc[4];
#pragma unroll
  for (int mt = 0; mt < 4; ++mt) outc[mt] = zero8();

  const int hwave = wv * 128;
#pragma unroll 1
  for (int c = 0; c < 4; ++c) {
    const int hc = hwave + c * 32;
    v8f hacc[4][2];
#pragma unroll
    for (int mt = 0; mt < 4; ++mt) { hacc[mt][0] = zero8(); hacc[mt][1] = zero8(); }

    // Keep this loop ROLLED: full unroll (16 k-steps) blows past 256 VGPRs
    // and forces scratch spills in the WMMA pipeline.
#pragma unroll 1
    for (int k0 = 0; k0 < D_IN; k0 += 32) {
      v16bf bf0 = ld16(w1e + (hc + m)      * D_IN + k0 + kg * 16, 8);
      v16bf bf1 = ld16(w1e + (hc + 16 + m) * D_IN + k0 + kg * 16, 8);
#pragma unroll
      for (int mt = 0; mt < 4; ++mt) {
        v16bf a = ld16(xrow[mt] + k0 + kg * 16, 8);
        hacc[mt][0] = __builtin_amdgcn_wmma_f32_16x16x32_bf16(
            false, a, false, bf0, (short)0, hacc[mt][0], false, false);
        hacc[mt][1] = __builtin_amdgcn_wmma_f32_16x16x32_bf16(
            false, a, false, bf1, (short)0, hacc[mt][1], false, false);
      }
    }

    v16bf w2f = ld16(w2e + m * H_DIM + hc + kg * 16, 8);
    float bia0 = b1[e * H_DIM + hc + m];
    float bia1 = b1[e * H_DIM + hc + 16 + m];

    // FULLY unrolled so hacc/outc/xrow stay statically indexed (a rolled loop
    // here turns them into scratch-backed arrays -> 16x scratch_store_b128).
#pragma unroll
    for (int mt = 0; mt < 4; ++mt) {
      // C layout: VGPR g holds (row = g + 8*kg, col = m). Write relu(h) as
      // bf16 into per-wave LDS in row-major [16][32], then reload in A layout.
#pragma unroll
      for (int g = 0; g < 8; ++g) {
        int row = g + 8 * kg;
        hs[wv][row * 32 + m]      = f2bf(fmaxf(hacc[mt][0][g] + bia0, 0.f));
        hs[wv][row * 32 + 16 + m] = f2bf(fmaxf(hacc[mt][1][g] + bia1, 0.f));
      }
      asm volatile("s_wait_dscnt 0x0" ::: "memory");  // wave-local store->load
      v16bf ha = ld16(&hs[wv][m * 32 + kg * 16], 8);
      outc[mt] = __builtin_amdgcn_wmma_f32_16x16x32_bf16(
          false, ha, false, w2f, (short)0, outc[mt], false, false);
    }
  }

  // cross-wave reduction of the 16x16 out tiles (each wave has a partial over
  // its 128 H columns)
#pragma unroll
  for (int mt = 0; mt < 4; ++mt)
#pragma unroll
    for (int g = 0; g < 8; ++g)
      red[((wv * 4 + mt) * 8 + g) * 32 + lane] = outc[mt][g];
  __syncthreads();

  const int rmt = wv & 3;
  const int gb  = (wv >> 2) * 4;
#pragma unroll
  for (int gi = 0; gi < 4; ++gi) {
    int g = gb + gi;
    float s = 0.f;
#pragma unroll
    for (int ww = 0; ww < 8; ++ww) s += red[((ww * 4 + rmt) * 8 + g) * 32 + lane];
    int   row = g + 8 * kg;
    int   tok = toks[rmt * 16 + row];
    float w   = gws[rmt * 16 + row];
    int   n   = m;
    if (n < O_DIM && w != 0.f)
      atomicAdd(&outW[tok * O_DIM + n], w * (s + b2[e * O_DIM + n]));
  }
}

// ---------------------------------------------------------------- launch
extern "C" void kernel_launch(void* const* d_in, const int* in_sizes, int n_in,
                              void* d_out, int out_size, void* d_ws, size_t ws_size,
                              hipStream_t stream) {
  const float* x  = (const float*)d_in[0];
  const float* W1 = (const float*)d_in[1];
  const float* b1 = (const float*)d_in[2];
  const float* W2 = (const float*)d_in[3];
  const float* b2 = (const float*)d_in[4];
  const float* Wg = (const float*)d_in[5];
  const float* bg = (const float*)d_in[6];

  float* outW     = (float*)d_out;              // [B,10]
  float* gatesOut = outW + B_TOK * O_DIM;       // [B,16]

  // workspace layout (~34.5 MiB)
  char* ws   = (char*)d_ws;
  u16*  xbf  = (u16*)(ws);                                  // 16 MiB
  u16*  w1t  = (u16*)(ws + (16u << 20));                    // 16 MiB
  u16*  w2t  = (u16*)(ws + (32u << 20));                    // 512 KiB
  int*  cnt  = (int*)(ws + (32u << 20) + (512u << 10));     // 64 B (pad 256)
  int*  lists = (int*)(ws + (32u << 20) + (512u << 10) + 256);        // 1 MiB
  float* gwbuf = (float*)((char*)lists + (size_t)N_EXP * B_TOK * 4);  // 1 MiB

  init_kernel<<<(B_TOK * O_DIM + 255) / 256, 256, 0, stream>>>(outW, cnt);
  cvt_x<<<(B_TOK * D_IN + 255) / 256, 256, 0, stream>>>(x, xbf);
  cvt_w1<<<(N_EXP * H_DIM * D_IN + 255) / 256, 256, 0, stream>>>(W1, w1t);
  cvt_w2<<<(N_EXP * 16 * H_DIM + 255) / 256, 256, 0, stream>>>(W2, w2t);
  gate_kernel<<<B_TOK / 8, 256, 0, stream>>>(x, Wg, bg, gatesOut, cnt, lists, gwbuf);

  dim3 grid(B_TOK / TPB_TOK, N_EXP);  // max tiles per expert x experts
  moe_kernel<<<grid, 256, 0, stream>>>(xbf, w1t, w2t, b1, b2, cnt, lists, gwbuf, outW);
}
